// Attention_28475633172551
// MI455X (gfx1250) — matrix operand order repaired and checked by reading, not compiled, not run on hardware
//
#include <hip/hip_runtime.h>

// ---------------------------------------------------------------------------
// Bilinear-recurrence RNN for MI455X (gfx1250).
//
// Per step: new_s = LN( einsum('bij,ijk->bk', s⊗s, P) + x_t Q^T ),
//           out_t = new_s R^T + x_t S^T
// The einsum is a GEMM  A[32 x 65536] * Pflat[65536 x 256]  with
// A[b, i*256+j] = s[b,i]*s[b,j]  (rank-1 structure -> A-tiles built in-regs).
//
// Design (latency-bound recurrence, 1024 serial steps):
//  * bf16 v_wmma_f32_16x16x32_bf16 for the quadratic contraction, f32 accum.
//  * P converted once per launch to bf16 in WMMA-B lane layout (32 MB,
//    L2-resident on the 192 MB L2; HBM is irrelevant for this kernel).
//  * Persistent grid of 16 WGs x 512 threads (16 waves; wave w owns N-tile w,
//    WG g owns K-slice [g*4096, (g+1)*4096)).  Small grid keeps the 3x1024
//    grid-barrier cost low; per-WG per-step math is ~67 MFLOP.
//  * LN + R/Q/S side-GEMMs in f32 VALU (tiny: ~1.2% of FLOPs).
//
// d_ws requirement: ~34.2 MB.
// ---------------------------------------------------------------------------

typedef __attribute__((ext_vector_type(16))) __bf16 v16bf;
typedef __attribute__((ext_vector_type(8)))  float  v8f;

#define NWG      16
#define THREADS  512
#define BATCH    32
#define DIM      256
#define SEQ      1024
#define CHUNKS   128            // k-chunks of 32 per WG  (16 i-rows * 256 j / 32)
#define KTILES   2048           // 65536 / 32 total k-chunks

// d_ws layout (bytes)
#define OFF_BAR     0ul                         // 2 x u32 barrier state
#define OFF_STATE   256ul                       // 32*256 f32 state
#define OFF_ROWSTAT 33280ul                     // 16*32*2 f32 LN partial stats
#define OFF_PART    37376ul                     // 16*32*256 f32 GEMM partials
#define OFF_PB      561664ul                    // 2048*16*32*16 bf16 packed P
// total = 561664 + 33554432 = 34116096 bytes

// ---------------------------------------------------------------------------
// Pack P (f32 [256,256,256], k fastest) into bf16 WMMA-B operand layout.
// B-tile (K=32 x N=16), contiguous k order used by this file: lanes 0-15 hold K=0..15 for N=lane,
// lanes 16-31 hold K=16..31 for N=lane-16; element e of the v16bf is K step e.
// B element (row K=ij, col N=n) = P[i][j][n] = Pflat[ij*256 + n].
// ---------------------------------------------------------------------------
__global__ void pack_P_bf16(const float* __restrict__ P, __bf16* __restrict__ Pb) {
    int tile = blockIdx.x;                 // kt*16 + nt
    int kt   = tile >> 4;
    int nt   = tile & 15;
    int lane = threadIdx.x >> 4;           // 0..31
    int e    = threadIdx.x & 15;           // 0..15
    int kin  = (lane & 16) + e;            // K within tile
    int n    = nt * 16 + (lane & 15);
    long ij  = (long)kt * 32 + kin;
    Pb[((long)tile * 32 + lane) * 16 + e] = (__bf16)P[ij * 256 + n];
}

__global__ void init_ws(float* __restrict__ state, unsigned* __restrict__ bar) {
    int t = blockIdx.x * blockDim.x + threadIdx.x;
    if (t < BATCH * DIM) state[t] = 0.f;
    if (t == 0) { bar[0] = 0u; bar[1] = 0u; }
}

// Sense-reversing grid barrier (agent scope). 16 co-resident WGs.
__device__ inline void grid_barrier(unsigned* cnt, unsigned* gen) {
    __threadfence();                        // make this WG's global writes visible
    __syncthreads();
    if (threadIdx.x == 0) {
        unsigned g = __hip_atomic_load(gen, __ATOMIC_ACQUIRE, __HIP_MEMORY_SCOPE_AGENT);
        unsigned a = __hip_atomic_fetch_add(cnt, 1u, __ATOMIC_ACQ_REL, __HIP_MEMORY_SCOPE_AGENT);
        if (a == (unsigned)(NWG - 1)) {
            __hip_atomic_store(cnt, 0u, __ATOMIC_RELAXED, __HIP_MEMORY_SCOPE_AGENT);
            __hip_atomic_fetch_add(gen, 1u, __ATOMIC_RELEASE, __HIP_MEMORY_SCOPE_AGENT);
        } else {
            while (__hip_atomic_load(gen, __ATOMIC_ACQUIRE, __HIP_MEMORY_SCOPE_AGENT) == g)
                __builtin_amdgcn_s_sleep(2);
        }
    }
    __syncthreads();
}

__global__ void __launch_bounds__(THREADS)
bilinear_rnn(const float* __restrict__ x,   const float* __restrict__ Q,
             const float* __restrict__ R,   const float* __restrict__ Smat,
             const float* __restrict__ lnw, const float* __restrict__ lnb,
             const __bf16* __restrict__ Pb,
             float* __restrict__ out, float* __restrict__ state,
             float* __restrict__ part, float* __restrict__ rowstat,
             unsigned* __restrict__ bar)
{
    __shared__ float sS[BATCH][DIM + 1];     // +1 pad: conflict-free strided reads
    __shared__ float preS[BATCH][16];
    __shared__ float muS[BATCH], rsS[BATCH];

    const int wg    = blockIdx.x;
    const int tid   = threadIdx.x;
    const int wave  = tid >> 5;              // N-tile id 0..15
    const int lane  = tid & 31;
    const int lhalf = lane >> 4;
    const int lmod  = lane & 15;

    // rows handled by the two M-tiles for this lane
    const int m0 = lmod;
    const int m1 = lmod + 16;

    // Phase B/C/D thread mapping
    const int bB   = tid >> 4;               // batch row 0..31
    const int nl   = tid & 15;
    const int ncol = wg * 16 + nl;           // output/state column owned here

    // initial state (zeros) -> LDS
    for (int i = tid; i < BATCH * DIM; i += THREADS)
        sS[i >> 8][i & 255] = state[i];
    __syncthreads();

    for (int t = 0; t < SEQ; ++t) {
        // ---- Phase A: WMMA partial GEMM over this WG's K-slice ------------
        v8f acc0 = {0.f,0.f,0.f,0.f,0.f,0.f,0.f,0.f};
        v8f acc1 = acc0;
        #pragma unroll 2
        for (int kk = 0; kk < CHUNKS; ++kk) {
            const __bf16* bp =
                Pb + ((((long)(wg * CHUNKS + kk) * 16 + wave) * 32 + lane) * 16);
            v16bf bt = *(const v16bf*)bp;
            __builtin_prefetch((const void*)(bp + 16 * 512), 0, 0); // next k-chunk

            int Ig = wg * 16 + (kk >> 3);    // global i of this chunk's A rows
            int j0 = (kk & 7) * 32;          // j base of this chunk
            float sI0 = sS[m0][Ig];
            float sI1 = sS[m1][Ig];
            v16bf a0, a1;
            #pragma unroll
            for (int e = 0; e < 16; ++e) {
                // A layout (matches packed B): lanes<16 K=0..15, lanes>=16 K=16..31
                int k = e + (lhalf << 4);
                float sj0 = sS[m0][j0 + k];
                float sj1 = sS[m1][j0 + k];
                a0[e] = (__bf16)(sI0 * sj0);
                a1[e] = (__bf16)(sI1 * sj1);
            }
            acc0 = __builtin_amdgcn_wmma_f32_16x16x32_bf16(
                       false, a0, false, bt, (short)0, acc0, false, false);
            acc1 = __builtin_amdgcn_wmma_f32_16x16x32_bf16(
                       false, a1, false, bt, (short)0, acc1, false, false);
        }
        // C layout: VGPR r -> row r + 8*lhalf, col = lane&15 (per ISA table)
        {
            float* pbase = part + (long)wg * BATCH * DIM + wave * 16 + lmod;
            #pragma unroll
            for (int r = 0; r < 8; ++r) {
                pbase[(r + 8 * lhalf) * DIM]      = acc0[r];
                pbase[(r + 8 * lhalf + 16) * DIM] = acc1[r];
            }
        }
        grid_barrier(bar, bar + 1);

        // ---- Phase B: reduce K-partials + Q x_t; per-row LN partial stats --
        {
            float v = 0.f;
            for (int w = 0; w < NWG; ++w)
                v += part[((long)w * BATCH + bB) * DIM + ncol];
            const float* xr = x + ((long)bB * SEQ + t) * DIM;
            const float* qr = Q + (long)ncol * DIM;
            float qx = 0.f;
            for (int i2 = 0; i2 < DIM; ++i2) qx += xr[i2] * qr[i2];
            preS[bB][nl] = v + qx;
        }
        __syncthreads();
        if (tid < BATCH) {
            float s1 = 0.f, s2 = 0.f;
            #pragma unroll
            for (int c = 0; c < 16; ++c) { float u = preS[tid][c]; s1 += u; s2 += u * u; }
            rowstat[((long)wg * BATCH + tid) * 2 + 0] = s1;
            rowstat[((long)wg * BATCH + tid) * 2 + 1] = s2;
        }
        grid_barrier(bar, bar + 1);

        // ---- Phase C: LayerNorm, write new state --------------------------
        if (tid < BATCH) {
            float s1 = 0.f, s2 = 0.f;
            for (int g2 = 0; g2 < NWG; ++g2) {
                s1 += rowstat[((long)g2 * BATCH + tid) * 2 + 0];
                s2 += rowstat[((long)g2 * BATCH + tid) * 2 + 1];
            }
            float mu  = s1 * (1.f / DIM);
            float var = s2 * (1.f / DIM) - mu * mu;
            muS[tid] = mu;
            rsS[tid] = rsqrtf(var + 1e-5f);
        }
        __syncthreads();
        {
            float v  = preS[bB][nl];
            float sn = (v - muS[bB]) * rsS[bB] * lnw[ncol] + lnb[ncol];
            state[bB * DIM + ncol] = sn;
        }
        grid_barrier(bar, bar + 1);

        // ---- Phase D: reload full new state; emit output ------------------
        for (int i = tid; i < BATCH * DIM; i += THREADS)
            sS[i >> 8][i & 255] = state[i];
        __syncthreads();
        {
            const float* xr = x + ((long)bB * SEQ + t) * DIM;
            const float* rr = R + (long)ncol * DIM;
            const float* sr = Smat + (long)ncol * DIM;
            float o = 0.f;
            for (int k2 = 0; k2 < DIM; ++k2) o += sS[bB][k2] * rr[k2];
            for (int i2 = 0; i2 < DIM; ++i2) o += xr[i2] * sr[i2];
            out[((long)bB * SEQ + t) * DIM + ncol] = o;
        }
        // sS already holds new state for next step's Phase A.
    }
}

extern "C" void kernel_launch(void* const* d_in, const int* in_sizes, int n_in,
                              void* d_out, int out_size, void* d_ws, size_t ws_size,
                              hipStream_t stream) {
    (void)in_sizes; (void)n_in; (void)out_size; (void)ws_size;
    const float* x   = (const float*)d_in[0];
    const float* P   = (const float*)d_in[1];
    const float* Q   = (const float*)d_in[2];
    const float* R   = (const float*)d_in[3];
    const float* S   = (const float*)d_in[4];
    const float* lnw = (const float*)d_in[5];
    const float* lnb = (const float*)d_in[6];
    float* out = (float*)d_out;

    char* ws = (char*)d_ws;
    unsigned* bar   = (unsigned*)(ws + OFF_BAR);
    float* state    = (float*)(ws + OFF_STATE);
    float* rowstat  = (float*)(ws + OFF_ROWSTAT);
    float* part     = (float*)(ws + OFF_PART);
    __bf16* Pb      = (__bf16*)(ws + OFF_PB);

    init_ws<<<32, 256, 0, stream>>>(state, bar);
    pack_P_bf16<<<KTILES * 16, 512, 0, stream>>>(P, Pb);
    bilinear_rnn<<<NWG, THREADS, 0, stream>>>(x, Q, R, S, lnw, lnb, Pb,
                                              out, state, part, rowstat, bar);
}
